// UnitY2AlignmentEncoder_72000831750218
// MI455X (gfx1250) — compile-verified, matrix operand order checked by reading, not run on hardware
//
#include <hip/hip_runtime.h>
#include <hip/hip_bf16.h>
#include <math.h>

// ---------------------------------------------------------------------------
// Problem constants (from the reference)
// ---------------------------------------------------------------------------
#define BB   8
#define TT   1024      // text length
#define TF   4096      // feature length
#define TFR  2048      // reduced feature length (stride R=2)
#define CC   256       // embed dim
#define FD   80        // feat input dim
#define FDP  96        // feat input dim padded to multiple of 32

typedef __attribute__((ext_vector_type(16))) _Float16 v16h;
typedef __attribute__((ext_vector_type(8)))  float    v8f;

union Frag16 { v16h v; unsigned u[8]; };

// Load a 16-half WMMA A/B fragment for row-major [row][c] f16 data.
// CDNA5 16-bit A layout: lane = M + 16*g ; VGPR v holds K pair
//   K = ((v<4)?0:16) + g*8 + (v&3)*2   -> dword index = ((v<4)?0:8) + g*4 + (v&3)
// B (K x N) uses the symmetric layout with lane = N + 16*g.
__device__ __forceinline__ v16h load_frag(const _Float16* rowptr, int cbase,
                                          int g, bool valid) {
  Frag16 f;
  const unsigned* p = reinterpret_cast<const unsigned*>(rowptr + cbase);
#pragma unroll
  for (int v = 0; v < 8; ++v) {
    int ui = ((v < 4) ? 0 : 8) + g * 4 + (v & 3);
    f.u[v] = valid ? p[ui] : 0u;
  }
  return f.v;
}

// ---------------------------------------------------------------------------
// Prep kernels
// ---------------------------------------------------------------------------
__global__ void cvt_f32_to_f16_kernel(const float* __restrict__ x,
                                      _Float16* __restrict__ y, int n) {
  int i = blockIdx.x * blockDim.x + threadIdx.x;
  if (i < n) y[i] = (_Float16)x[i];
}

// feat_emb (B, TF, FD) fp32 -> (B, TF, FDP) f16 zero-padded
__global__ void cvt_feat_pad_kernel(const float* __restrict__ x,
                                    _Float16* __restrict__ y) {
  int i = blockIdx.x * blockDim.x + threadIdx.x;
  const int n = BB * TF * FDP;
  if (i >= n) return;
  int c = i % FDP;                          // constant modulo
  int bt = i / FDP;                         // b*TF + t
  float v = (c < FD) ? x[bt * FD + c] : 0.f;
  y[i] = (_Float16)v;
}

// w (Cout=256, CIN, K) fp32 -> wt [K][256][CPAD] f16 (ci zero-padded)
template <int CIN, int K, int CPAD>
__global__ void prep_weight_kernel(const float* __restrict__ w,
                                   _Float16* __restrict__ wt) {
  int i = blockIdx.x * blockDim.x + threadIdx.x;
  const int total = K * 256 * CPAD;
  if (i >= total) return;
  int ci = i % CPAD;
  int r  = i / CPAD;
  int co = r % 256;
  int k  = r / 256;
  float v = (ci < CIN) ? w[(co * CIN + ci) * K + k] : 0.f;
  wt[i] = (_Float16)v;
}

// ---------------------------------------------------------------------------
// Generic conv1d (same-layout GEMM) on WMMA, fully unrolled.
// out[b,t,co] = act( bias[co] + sum_{k,ci} in[b, t*STRIDE + k - PAD, ci] * W[k,co,ci] )
// One wave computes a 16(t) x 16(co) output tile.
// Grid: x = T_out/8 blocks (8 waves each -> T_out wave-tiles), y = batch.
// ---------------------------------------------------------------------------
template <int KTAPS, int CPAD, int STRIDE, int PAD, bool RELU>
__global__ void conv_wmma_kernel(const _Float16* __restrict__ inH,   // [B][T_in][CPAD]
                                 const _Float16* __restrict__ wH,    // [KTAPS][256][CPAD]
                                 const float*    __restrict__ bias,  // [256]
                                 _Float16*       __restrict__ outH,  // [B][T_out][256]
                                 int T_in, int T_out) {
  int lane = threadIdx.x & 31;
  int idx  = blockIdx.x * 8 + (threadIdx.x >> 5);   // tile idx within batch
  int b    = blockIdx.y;
  int ttile  = idx >> 4;                            // T_out/16 t-tiles
  int cotile = idx & 15;                            // 16 co-tiles (Cout=256)
  if (ttile >= (T_out >> 4)) return;

  int n = lane & 15, g = lane >> 4;
  int trowBase = (ttile * 16 + n) * STRIDE - PAD;   // A-lane row (M = lane&15)
  const _Float16* inB = inH + (size_t)b * T_in * CPAD;

  v8f acc = {};
#pragma unroll
  for (int k = 0; k < KTAPS; ++k) {
    int trow = trowBase + k;
    bool valid = (trow >= 0) && (trow < T_in);
    const _Float16* arow = inB + (size_t)(valid ? trow : 0) * CPAD;
    const _Float16* wrow = wH + (k * 256 + (cotile * 16 + n)) * CPAD;
#pragma unroll
    for (int cc = 0; cc < CPAD; cc += 32) {
      v16h a  = load_frag(arow, cc, g, valid);
      v16h bb = load_frag(wrow, cc, g, true);
      acc = __builtin_amdgcn_wmma_f32_16x16x32_f16(false, a, false, bb,
                                                   (short)0, acc, false, false);
    }
  }

  float bn = bias[cotile * 16 + n];
  size_t outBase = ((size_t)b * T_out + (size_t)ttile * 16) * 256 + cotile * 16 + n;
#pragma unroll
  for (int rr = 0; rr < 8; ++rr) {
    int m = rr + 8 * g;                    // D layout: lane n, row rr+8*g
    float v = acc[rr] + bn;
    if (RELU) v = fmaxf(v, 0.f);
    outH[outBase + (size_t)m * 256] = (_Float16)v;
  }
}

// ---------------------------------------------------------------------------
// Row sum-of-squares (f16 input, f32 out), one wave per row
// ---------------------------------------------------------------------------
__global__ void sumsq_kernel(const _Float16* __restrict__ x,
                             float* __restrict__ out, int rows) {
  int row = blockIdx.x * 8 + (threadIdx.x >> 5);
  int lane = threadIdx.x & 31;
  if (row >= rows) return;
  const _Float16* p = x + (size_t)row * CC;
  float s = 0.f;
#pragma unroll
  for (int c = 0; c < CC / 32; ++c) {
    float v = (float)p[lane + c * 32];
    s += v * v;
  }
  for (int o = 16; o > 0; o >>= 1) s += __shfl_xor(s, o, 32);
  if (lane == 0) out[row] = s;
}

// ---------------------------------------------------------------------------
// Distance score: score[b,fi,tj] = -sqrt(max(|f|^2+|t|^2-2 f.t, 1e-12)),
// masked to -inf for tj >= text_len[b].
// One wave owns one 16-row f-tile and 4 consecutive 16-col t-tiles; the A
// (feature) fragments for the whole K=256 stay in registers and are reused
// across the 4 t-tiles (4x less A traffic, 32 back-to-back WMMAs per wave).
// Grid: x = 256 blocks (128 ftiles * 16 t-groups / 8 waves), y = batch.
// ---------------------------------------------------------------------------
__global__ void dist_score_kernel(const _Float16* __restrict__ fh,  // [B][TFR][256]
                                  const _Float16* __restrict__ th,  // [B][TT][256]
                                  const float* __restrict__ sf,     // [B*TFR]
                                  const float* __restrict__ st,     // [B*TT]
                                  const int*   __restrict__ tlen,
                                  float* __restrict__ score) {      // [B][TFR][TT]
  int lane = threadIdx.x & 31;
  int idx  = blockIdx.x * 8 + (threadIdx.x >> 5);   // 0 .. 128*16-1
  int b    = blockIdx.y;
  int ftile = idx >> 4;                             // 128 f-tiles
  int tgrp  = idx & 15;                             // 16 groups of 4 t-tiles

  int n = lane & 15, g = lane >> 4;

  // A fragments: whole K=256 for this wave's 16 feature rows.
  const _Float16* arow = fh + ((size_t)b * TFR + ftile * 16 + n) * 256;
  v16h afrag[8];
#pragma unroll
  for (int cc = 0; cc < 8; ++cc) afrag[cc] = load_frag(arow, cc * 32, g, true);

  // |f|^2 for the 8 rows this lane writes.
  float sfv[8];
#pragma unroll
  for (int rr = 0; rr < 8; ++rr)
    sfv[rr] = sf[(size_t)b * TFR + ftile * 16 + rr + 8 * g];

  int TL = tlen[b];

#pragma unroll
  for (int tt = 0; tt < 4; ++tt) {
    int ttile = tgrp * 4 + tt;
    const _Float16* brow = th + ((size_t)b * TT + ttile * 16 + n) * 256;
    v8f acc = {};
#pragma unroll
    for (int cc = 0; cc < 8; ++cc) {
      v16h bb = load_frag(brow, cc * 32, g, true);
      acc = __builtin_amdgcn_wmma_f32_16x16x32_f16(false, afrag[cc], false, bb,
                                                   (short)0, acc, false, false);
    }

    int tj = ttile * 16 + n;
    float stn = st[(size_t)b * TT + tj];
    bool masked = (tj >= TL);
#pragma unroll
    for (int rr = 0; rr < 8; ++rr) {
      int fi = ftile * 16 + rr + 8 * g;
      float d2 = sfv[rr] + stn - 2.f * acc[rr];
      float s = -sqrtf(fmaxf(d2, 1e-12f));
      if (masked) s = -INFINITY;
      score[((size_t)b * TFR + fi) * TT + tj] = s;
    }
  }
}

// ---------------------------------------------------------------------------
// In-place row log-softmax over the last (TT) axis; one block per row.
// ---------------------------------------------------------------------------
__global__ void log_softmax_kernel(float* __restrict__ x) {
  int row = blockIdx.x;
  float* p = x + (size_t)row * TT;
  __shared__ float red[8];
  __shared__ float bval;
  int tid = threadIdx.x, lane = tid & 31, w = tid >> 5;

  float m = -INFINITY;
  for (int i = tid; i < TT; i += 256) m = fmaxf(m, p[i]);
  for (int o = 16; o > 0; o >>= 1) m = fmaxf(m, __shfl_xor(m, o, 32));
  if (lane == 0) red[w] = m;
  __syncthreads();
  if (tid == 0) {
    float v = red[0];
    for (int k = 1; k < 8; ++k) v = fmaxf(v, red[k]);
    bval = v;
  }
  __syncthreads();
  m = bval;
  __syncthreads();

  float s = 0.f;
  for (int i = tid; i < TT; i += 256) s += expf(p[i] - m);
  for (int o = 16; o > 0; o >>= 1) s += __shfl_xor(s, o, 32);
  if (lane == 0) red[w] = s;
  __syncthreads();
  if (tid == 0) {
    float v = 0.f;
    for (int k = 0; k < 8; ++k) v += red[k];
    bval = logf(v);
  }
  __syncthreads();
  float lse = bval;

  for (int i = tid; i < TT; i += 256) p[i] = p[i] - m - lse;
}

// ---------------------------------------------------------------------------
// MAS forward: Q[j,i] = max(Q[j-1,i-1], Q[j-1,i]) + lp[j,i].
// One block (1024 threads) per batch, running column in LDS, full Q to ws.
// ---------------------------------------------------------------------------
__global__ void mas_forward_kernel(const float* __restrict__ lp,
                                   float* __restrict__ Q) {
  int b = blockIdx.x;
  int i = threadIdx.x;                     // 0..1023
  __shared__ float qP[TT];
  const float* lpb = lp + (size_t)b * TFR * TT;
  float* Qb = Q + (size_t)b * TFR * TT;

  float q = (i == 0) ? lpb[0] : -INFINITY;
  qP[i] = q;
  Qb[i] = q;
  __syncthreads();

  for (int j = 1; j < TFR; ++j) {
    float qm1 = (i > 0) ? qP[i - 1] : -INFINITY;
    float q0 = qP[i];
    __syncthreads();
    float qn = fmaxf(qm1, q0) + lpb[(size_t)j * TT + i];
    qP[i] = qn;
    Qb[(size_t)j * TT + i] = qn;
    __syncthreads();
  }
}

// ---------------------------------------------------------------------------
// MAS backtrack + duration postprocess; one block per batch (scalar logic).
// ---------------------------------------------------------------------------
__global__ void mas_backtrack_kernel(const float* __restrict__ Q,
                                     const int* __restrict__ tlen,
                                     const int* __restrict__ flen,
                                     float* __restrict__ dur_out) {
  int b = blockIdx.x;
  __shared__ int cnt[TT];
  for (int i = threadIdx.x; i < TT; i += blockDim.x) cnt[i] = 0;
  __syncthreads();

  if (threadIdx.x == 0) {
    int tl = tlen[b];
    int flfull = flen[b];
    int fl = (flfull + 1) >> 1;            // ceil(feat_len / R), R = 2
    const float* Qb = Q + (size_t)b * TFR * TT;

    int a = tl - 1;
    if (TFR - 1 < fl) cnt[a]++;            // A[TFR-1] = tl-1
    for (int j = TFR - 2; j >= 0; --j) {
      const float* qc = Qb + (size_t)j * TT;
      int cand = (a == 0) ? 0 : ((qc[a - 1] >= qc[a]) ? (a - 1) : a);
      a = (j >= fl - 1) ? (tl - 1) : cand;
      if (j < fl) cnt[a]++;
    }

    // postprocess: dur *= R, truncate at first token whose cumsum >= feat_len
    long csum = 0, prevcs = 0;
    int found = 0, tstar = -1;
    for (int i = 0; i < TT; ++i) {
      long d = 2L * cnt[i];
      csum += d;
      if (!found && csum >= flfull && i < tl) { found = 1; tstar = i; prevcs = csum - d; }
    }
    for (int i = 0; i < TT; ++i) {
      float v = (float)(2 * cnt[i]);
      if (found) {
        if (i == tstar) v = (float)(flfull - prevcs);
        else if (i > tstar) v = 0.f;
      }
      dur_out[(size_t)b * TT + i] = v;
    }
  }
}

// ---------------------------------------------------------------------------
// Host launch
// ---------------------------------------------------------------------------
extern "C" void kernel_launch(void* const* d_in, const int* in_sizes, int n_in,
                              void* d_out, int out_size, void* d_ws, size_t ws_size,
                              hipStream_t stream) {
  (void)in_sizes; (void)n_in; (void)out_size; (void)ws_size;

  const float* text_emb = (const float*)d_in[0];   // (8,1024,256)
  const float* feat_emb = (const float*)d_in[1];   // (8,4096,80)
  const int*   tlen     = (const int*)d_in[2];
  const int*   flen     = (const int*)d_in[3];
  const float* tw0 = (const float*)d_in[4];  const float* tb0 = (const float*)d_in[5];
  const float* tw1 = (const float*)d_in[6];  const float* tb1 = (const float*)d_in[7];
  const float* fw0 = (const float*)d_in[8];  const float* fb0 = (const float*)d_in[9];
  const float* fw1 = (const float*)d_in[10]; const float* fb1 = (const float*)d_in[11];
  const float* fw2 = (const float*)d_in[12]; const float* fb2 = (const float*)d_in[13];

  char* ws = (char*)d_ws;
  size_t off = 0;
  auto alloc = [&](size_t bytes) -> char* {
    char* p = ws + off;
    off = (off + bytes + 255) & ~(size_t)255;
    return p;
  };

  _Float16* th0   = (_Float16*)alloc((size_t)BB * TT  * CC  * 2);  // text f16
  _Float16* xh0   = (_Float16*)alloc((size_t)BB * TF  * FDP * 2);  // feat f16 padded
  _Float16* t1h   = (_Float16*)alloc((size_t)BB * TT  * CC  * 2);
  _Float16* t2h   = (_Float16*)alloc((size_t)BB * TT  * CC  * 2);  // final text emb
  _Float16* y1h   = (_Float16*)alloc((size_t)BB * TF  * CC  * 2);
  _Float16* y2h   = (_Float16*)alloc((size_t)BB * TF  * CC  * 2);
  _Float16* f3h   = (_Float16*)alloc((size_t)BB * TFR * CC  * 2);  // final feat emb
  _Float16* wtT0  = (_Float16*)alloc((size_t)3 * 256 * CC  * 2);
  _Float16* wtT1  = (_Float16*)alloc((size_t)1 * 256 * CC  * 2);
  _Float16* wtF0  = (_Float16*)alloc((size_t)3 * 256 * FDP * 2);
  _Float16* wtF1  = (_Float16*)alloc((size_t)3 * 256 * CC  * 2);
  _Float16* wtF2  = (_Float16*)alloc((size_t)1 * 256 * CC  * 2);
  float*    sf    = (float*)alloc((size_t)BB * TFR * 4);
  float*    st    = (float*)alloc((size_t)BB * TT  * 4);
  float*    Qws   = (float*)alloc((size_t)BB * TFR * TT * 4);

  float* attn = (float*)d_out;                                // (8,2048,1024)
  float* dur  = (float*)d_out + (size_t)BB * TFR * TT;        // (8,1024)

  // --- prep: f32 -> f16 conversions, weight repack -------------------------
  {
    int n = BB * TT * CC;
    cvt_f32_to_f16_kernel<<<(n + 255) / 256, 256, 0, stream>>>(text_emb, th0, n);
  }
  {
    int n = BB * TF * FDP;
    cvt_feat_pad_kernel<<<(n + 255) / 256, 256, 0, stream>>>(feat_emb, xh0);
  }
  prep_weight_kernel<CC, 3, CC><<<(3 * 256 * CC + 255) / 256, 256, 0, stream>>>(tw0, wtT0);
  prep_weight_kernel<CC, 1, CC><<<(1 * 256 * CC + 255) / 256, 256, 0, stream>>>(tw1, wtT1);
  prep_weight_kernel<FD, 3, FDP><<<(3 * 256 * FDP + 255) / 256, 256, 0, stream>>>(fw0, wtF0);
  prep_weight_kernel<CC, 3, CC><<<(3 * 256 * CC + 255) / 256, 256, 0, stream>>>(fw1, wtF1);
  prep_weight_kernel<CC, 1, CC><<<(1 * 256 * CC + 255) / 256, 256, 0, stream>>>(fw2, wtF2);

  // --- conv stacks (WMMA), grid.x = T_out/8 wave-tiles, grid.y = batch ----
  // text: conv(tw0,k3,p1,relu) -> conv(tw1,k1)
  conv_wmma_kernel<3, CC, 1, 1, true><<<dim3(TT / 8, BB), 256, 0, stream>>>(
      th0, wtT0, tb0, t1h, TT, TT);
  conv_wmma_kernel<1, CC, 1, 0, false><<<dim3(TT / 8, BB), 256, 0, stream>>>(
      t1h, wtT1, tb1, t2h, TT, TT);
  // feat: conv(fw0,k3,p1,relu) -> conv(fw1,k3,p1,relu) -> conv(fw2,k1,stride2)
  conv_wmma_kernel<3, FDP, 1, 1, true><<<dim3(TF / 8, BB), 256, 0, stream>>>(
      xh0, wtF0, fb0, y1h, TF, TF);
  conv_wmma_kernel<3, CC, 1, 1, true><<<dim3(TF / 8, BB), 256, 0, stream>>>(
      y1h, wtF1, fb1, y2h, TF, TF);
  conv_wmma_kernel<1, CC, 2, 0, false><<<dim3(TFR / 8, BB), 256, 0, stream>>>(
      y2h, wtF2, fb2, f3h, TF, TFR);

  // --- row norms -----------------------------------------------------------
  sumsq_kernel<<<(BB * TFR) / 8, 256, 0, stream>>>(f3h, sf, BB * TFR);
  sumsq_kernel<<<(BB * TT) / 8, 256, 0, stream>>>(t2h, st, BB * TT);

  // --- distance scores (WMMA) into d_out, then log-softmax ---------------
  // waves per batch = 128 ftiles * 16 t-groups = 2048 -> 256 blocks
  dist_score_kernel<<<dim3(256, BB), 256, 0, stream>>>(f3h, t2h, sf, st, tlen, attn);
  log_softmax_kernel<<<BB * TFR, 256, 0, stream>>>(attn);

  // --- MAS -----------------------------------------------------------------
  mas_forward_kernel<<<BB, TT, 0, stream>>>(attn, Qws);
  mas_backtrack_kernel<<<BB, 256, 0, stream>>>(Qws, tlen, flen, dur);
}
